// MultiHeadAttention_2946347565220
// MI455X (gfx1250) — compile-verified, matrix operand order checked by reading, not run on hardware
//
#include <hip/hip_runtime.h>

typedef __attribute__((ext_vector_type(16))) __bf16 v16bf;
typedef __attribute__((ext_vector_type(8)))  float  v8f;

union Frag {
    v16bf v;
    uint4 q[2];
    unsigned int   u[8];
    unsigned short s[16];
};

__device__ __forceinline__ unsigned int f2bf_bits(float f) {
    unsigned int u = __float_as_uint(f);
    return (u + 0x7FFFu + ((u >> 16) & 1u)) >> 16;   // RNE
}
__device__ __forceinline__ unsigned int pack2bf(float lo, float hi) {
    return f2bf_bits(lo) | (f2bf_bits(hi) << 16);
}
__device__ __forceinline__ float bias_val(const float* __restrict__ emb, int i, int j) {
    int idx = ((i < 64) == (j < 64)) ? (i > j ? i - j : j - i) : 1;
    return emb[idx];
}

// ---------------------------------------------------------------------------
// embedding output: [512 copies][256][256] of bias(i,j); pure streaming writes
// ---------------------------------------------------------------------------
__global__ __launch_bounds__(256)
void emb_bcast(const float* __restrict__ emb, float* __restrict__ out) {
    size_t gtid = (size_t)blockIdx.x * 256 + threadIdx.x;
    const size_t nthreads = (size_t)4096 * 256;
    #pragma unroll
    for (int j = 0; j < 8; ++j) {
        size_t idx4 = gtid + nthreads * j;       // float4 index, < 8388608
        size_t f = idx4 * 4;
        int i  = (int)((f >> 8) & 255);
        int c0 = (int)(f & 255);
        float4 v;
        v.x = bias_val(emb, i, c0 + 0);
        v.y = bias_val(emb, i, c0 + 1);
        v.z = bias_val(emb, i, c0 + 2);
        v.w = bias_val(emb, i, c0 + 3);
        ((float4*)out)[idx4] = v;
    }
}

// ---------------------------------------------------------------------------
// GEMM: C[M=8192, N=1024] = A[M,1024] @ W[1024,1024] + bvec
// Block tile 128x128, K-step 32, 8 waves x (32x64) each = 8 WMMA / 12 ds_load.
// Register-staged ping-pong LDS double buffering: global fetch of tile k+1
// overlaps WMMA on tile k; single barrier per K-step.
// MODE 0: bf16 out [b,h,s,d]; MODE 2: bf16 out [b,h,d,s]; MODE 1: f32 row-major
// ---------------------------------------------------------------------------
template<int ABF16, int MODE>
__global__ __launch_bounds__(256)
void gemm128x128(const void* __restrict__ Ain, const float* __restrict__ W,
                 const float* __restrict__ bvec, void* __restrict__ out) {
    __shared__ unsigned short As[2][128 * 40];  // row stride 40 bf16 (80B): conflict-free
    __shared__ unsigned short Bs[2][128 * 40];  // transposed: Bt[n][k]
    const int tid = threadIdx.x;
    const int w = tid >> 5, lane = tid & 31;
    const int lr = lane & 15, lh = lane >> 4;
    const int wr = w >> 1, wc = w & 1;          // wave: rows wr*32.., cols wc*64..
    const int m0 = blockIdx.y * 128;
    const int bn = blockIdx.x * 128;

    v8f acc[2][4];
    #pragma unroll
    for (int i = 0; i < 2; ++i)
        #pragma unroll
        for (int j = 0; j < 4; ++j) acc[i][j] = 0.0f;

    // staging registers
    float4 aSt[4];
    uint2  aSt16[4];
    float4 bSt[4];

    // ---- fetch(k0): global -> regs ------------------------------------
    auto fetch = [&](int k0) {
        #pragma unroll
        for (int j = 0; j < 4; ++j) {           // A tile: 128x32
            int i4 = tid + 256 * j;
            int r = i4 >> 3, c4 = (i4 & 7) << 2;
            if (ABF16) {
                aSt16[j] = *(const uint2*)
                    &((const unsigned short*)Ain)[(size_t)(m0 + r) * 1024 + k0 + c4];
            } else {
                aSt[j] = *(const float4*)
                    &((const float*)Ain)[(size_t)(m0 + r) * 1024 + k0 + c4];
            }
        }
        #pragma unroll
        for (int j = 0; j < 4; ++j) {           // B tile: 32x128
            int i4 = tid + 256 * j;
            int kr = i4 >> 5, c4 = (i4 & 31) << 2;
            bSt[j] = *(const float4*)&W[(size_t)(k0 + kr) * 1024 + bn + c4];
        }
    };
    // ---- store(p): regs -> LDS (A converted, B transposed) ------------
    auto store = [&](int p) {
        #pragma unroll
        for (int j = 0; j < 4; ++j) {
            int i4 = tid + 256 * j;
            int r = i4 >> 3, c4 = (i4 & 7) << 2;
            if (ABF16) {
                *(uint2*)&As[p][r * 40 + c4] = aSt16[j];
            } else {
                uint2 pk;
                pk.x = pack2bf(aSt[j].x, aSt[j].y);
                pk.y = pack2bf(aSt[j].z, aSt[j].w);
                *(uint2*)&As[p][r * 40 + c4] = pk;
            }
        }
        #pragma unroll
        for (int j = 0; j < 4; ++j) {
            int i4 = tid + 256 * j;
            int kr = i4 >> 5, c4 = (i4 & 31) << 2;
            Bs[p][(c4 + 0) * 40 + kr] = (unsigned short)f2bf_bits(bSt[j].x);
            Bs[p][(c4 + 1) * 40 + kr] = (unsigned short)f2bf_bits(bSt[j].y);
            Bs[p][(c4 + 2) * 40 + kr] = (unsigned short)f2bf_bits(bSt[j].z);
            Bs[p][(c4 + 3) * 40 + kr] = (unsigned short)f2bf_bits(bSt[j].w);
        }
    };

    fetch(0);
    int p = 0;
    for (int k0 = 0; k0 < 1024; k0 += 32) {
        store(p);
        __syncthreads();
        if (k0 + 32 < 1024) fetch(k0 + 32);     // overlaps WMMA below

        Frag a[2];
        #pragma unroll
        for (int tm = 0; tm < 2; ++tm) {
            const unsigned short* ap = &As[p][(wr * 32 + tm * 16 + lr) * 40 + lh * 16];
            a[tm].q[0] = *(const uint4*)ap;
            a[tm].q[1] = *(const uint4*)(ap + 8);
        }
        #pragma unroll
        for (int tn = 0; tn < 4; ++tn) {
            Frag b;
            const unsigned short* bp = &Bs[p][(wc * 64 + tn * 16 + lr) * 40 + lh * 16];
            b.q[0] = *(const uint4*)bp;
            b.q[1] = *(const uint4*)(bp + 8);
            #pragma unroll
            for (int tm = 0; tm < 2; ++tm)
                acc[tm][tn] = __builtin_amdgcn_wmma_f32_16x16x32_bf16(
                    false, a[tm].v, false, b.v, (short)0, acc[tm][tn], false, false);
        }
        p ^= 1;
    }

    // ---- epilogue
    #pragma unroll
    for (int tn = 0; tn < 4; ++tn) {
        int n = bn + wc * 64 + tn * 16 + lr;
        float bv = bvec[n];
        #pragma unroll
        for (int tm = 0; tm < 2; ++tm) {
            #pragma unroll
            for (int e = 0; e < 8; ++e) {
                int m = m0 + wr * 32 + tm * 16 + lh * 8 + e;
                float val = acc[tm][tn][e] + bv;
                if (MODE == 1) {
                    ((float*)out)[(size_t)m * 1024 + n] = val;
                } else {
                    int b = m >> 8, s = m & 255, h = n >> 6, d = n & 63;
                    size_t addr = (MODE == 0)
                        ? ((size_t)(b * 16 + h) * 256 + s) * 64 + d     // [b,h,s,d]
                        : ((size_t)(b * 16 + h) * 64 + d) * 256 + s;    // [b,h,d,s]
                    ((unsigned short*)out)[addr] = (unsigned short)f2bf_bits(val);
                }
            }
        }
    }
}

// ---------------------------------------------------------------------------
// attention: one block per (b,h, 32-row q-tile).
// scores = QK^T*scale + bias, mask -> softmax -> att (f32 out) -> ctx = att@V
// ---------------------------------------------------------------------------
__global__ __launch_bounds__(256)
void attn32(const unsigned short* __restrict__ qh,
            const unsigned short* __restrict__ kh,
            const unsigned short* __restrict__ vt,
            const int* __restrict__ mask,
            const float* __restrict__ emb,
            float* __restrict__ att,
            unsigned short* __restrict__ ctx) {
    __shared__ float sc[32 * 260];     // row stride 260 f32: conflict-free, 16B rows
    __shared__ float keepk[256];
    const int tid = threadIdx.x;
    const int w = tid >> 5, lane = tid & 31;
    const int lr = lane & 15, lh = lane >> 4;
    const int q0 = blockIdx.x * 32;
    const int bh = blockIdx.y;
    const int b = bh >> 4, h = bh & 15;
    const size_t base = (size_t)bh * 16384;    // per-(b,h) 256x64 bf16 tile

    keepk[tid] = mask[b * 256 + tid] ? 1.0f : 0.0f;
    __syncthreads();

    // ---- Q @ K^T: wave w owns tile-row tm (16 q rows), 4 tile-cols
    const int tm = w >> 2, tnb = (w & 3) * 4;
    v8f acc[4];
    #pragma unroll
    for (int i = 0; i < 4; ++i) acc[i] = 0.0f;

    #pragma unroll
    for (int kk = 0; kk < 64; kk += 32) {
        Frag a;
        const unsigned short* qp =
            &qh[base + (size_t)(q0 + tm * 16 + lr) * 64 + kk + lh * 16];
        a.q[0] = *(const uint4*)qp;
        a.q[1] = *(const uint4*)(qp + 8);
        #pragma unroll
        for (int i = 0; i < 4; ++i) {
            Frag bf;
            const unsigned short* kp =
                &kh[base + (size_t)((tnb + i) * 16 + lr) * 64 + kk + lh * 16];
            bf.q[0] = *(const uint4*)kp;
            bf.q[1] = *(const uint4*)(kp + 8);
            acc[i] = __builtin_amdgcn_wmma_f32_16x16x32_bf16(
                false, a.v, false, bf.v, (short)0, acc[i], false, false);
        }
    }

    // ---- scale + bias + mask -> LDS scores
    float kqv[8];
    #pragma unroll
    for (int e = 0; e < 8; ++e)
        kqv[e] = mask[b * 256 + q0 + tm * 16 + lh * 8 + e] ? 1.0f : 0.0f;
    #pragma unroll
    for (int i = 0; i < 4; ++i) {
        int sk = (tnb + i) * 16 + lr;
        float kkv = keepk[sk];
        #pragma unroll
        for (int e = 0; e < 8; ++e) {
            int sql = tm * 16 + lh * 8 + e;
            int sq = q0 + sql;
            float bv = bias_val(emb, sq, sk);
            sc[sql * 260 + sk] = (kqv[e] * kkv > 0.0f)
                ? acc[i][e] * 0.125f + bv : -1.0e9f;
        }
    }
    __syncthreads();

    // ---- row softmax (wave per 4 rows), apply keep, write back
    #pragma unroll
    for (int rr = 0; rr < 4; ++rr) {
        int r = w * 4 + rr;
        float v[8];
        float mx = -3.0e38f;
        #pragma unroll
        for (int j = 0; j < 8; ++j) {
            v[j] = sc[r * 260 + lane + 32 * j];
            mx = fmaxf(mx, v[j]);
        }
        for (int off = 16; off > 0; off >>= 1) mx = fmaxf(mx, __shfl_xor(mx, off, 32));
        float s = 0.0f;
        #pragma unroll
        for (int j = 0; j < 8; ++j) { v[j] = __expf(v[j] - mx); s += v[j]; }
        for (int off = 16; off > 0; off >>= 1) s += __shfl_xor(s, off, 32);
        float kq = mask[b * 256 + q0 + r] ? 1.0f : 0.0f;
        float inv = kq / fmaxf(s, 1e-20f);
        #pragma unroll
        for (int j = 0; j < 8; ++j)
            sc[r * 260 + lane + 32 * j] = v[j] * inv * keepk[lane + 32 * j];
    }
    __syncthreads();

    // ---- stream att tile to global (coalesced)
    float* attb = att + (size_t)bh * 65536 + (size_t)q0 * 256;
    for (int idx = tid; idx < 32 * 256; idx += 256) {
        int r = idx >> 8, c = idx & 255;
        attb[r * 256 + c] = sc[r * 260 + c];
    }

    // ---- ctx = att @ V : wave w owns one 16x16 output tile
    const int tm2 = w >> 2, tn2 = w & 3;
    v8f o = 0.0f;
    #pragma unroll
    for (int kk = 0; kk < 256; kk += 32) {
        Frag a;
        const float* srow = &sc[(tm2 * 16 + lr) * 260 + kk + lh * 16];
        #pragma unroll
        for (int t = 0; t < 4; ++t) {
            float4 f4 = *(const float4*)(srow + t * 4);
            a.u[t * 2 + 0] = pack2bf(f4.x, f4.y);
            a.u[t * 2 + 1] = pack2bf(f4.z, f4.w);
        }
        Frag bf;
        const unsigned short* vp =
            &vt[base + (size_t)(tn2 * 16 + lr) * 256 + kk + lh * 16];
        bf.q[0] = *(const uint4*)vp;
        bf.q[1] = *(const uint4*)(vp + 8);
        o = __builtin_amdgcn_wmma_f32_16x16x32_bf16(
            false, a.v, false, bf.v, (short)0, o, false, false);
    }
    #pragma unroll
    for (int e = 0; e < 8; ++e) {
        int sq = q0 + tm2 * 16 + lh * 8 + e;
        int d = tn2 * 16 + lr;
        ctx[(size_t)(b * 256 + sq) * 1024 + h * 64 + d] =
            (unsigned short)f2bf_bits(o[e]);
    }
}

// ---------------------------------------------------------------------------
extern "C" void kernel_launch(void* const* d_in, const int* in_sizes, int n_in,
                              void* d_out, int out_size, void* d_ws, size_t ws_size,
                              hipStream_t stream) {
    const float* q    = (const float*)d_in[0];
    const float* k    = (const float*)d_in[1];
    const float* v    = (const float*)d_in[2];
    const int*   mask = (const int*)d_in[4];
    const float* Wq = (const float*)d_in[5];  const float* bq = (const float*)d_in[6];
    const float* Wk = (const float*)d_in[7];  const float* bk = (const float*)d_in[8];
    const float* Wv = (const float*)d_in[9];  const float* bv = (const float*)d_in[10];
    const float* Wo = (const float*)d_in[11]; const float* bo = (const float*)d_in[12];
    const float* emb = (const float*)d_in[13];

    float* out     = (float*)d_out;                       // [32,256,1024]
    float* out_emb = out + (size_t)8388608;               // [32,16,256,256]
    float* out_att = out_emb + (size_t)33554432;          // [32,16,256,256]

    unsigned short* qh  = (unsigned short*)d_ws;          // bf16 [b,h,s,d]
    unsigned short* kh  = qh  + (size_t)8388608;          // bf16 [b,h,s,d]
    unsigned short* vt  = kh  + (size_t)8388608;          // bf16 [b,h,d,s] (V^T)
    unsigned short* ctx = vt  + (size_t)8388608;          // bf16 [b,s,h*64+d]

    emb_bcast<<<4096, 256, 0, stream>>>(emb, out_emb);

    dim3 gg(8, 64);   // N/128 x M/128
    gemm128x128<0, 0><<<gg, 256, 0, stream>>>(q, Wq, bq, qh);
    gemm128x128<0, 0><<<gg, 256, 0, stream>>>(k, Wk, bk, kh);
    gemm128x128<0, 2><<<gg, 256, 0, stream>>>(v, Wv, bv, vt);

    attn32<<<dim3(8, 512), 256, 0, stream>>>(qh, kh, vt, mask, emb, out_att, ctx);

    gemm128x128<1, 1><<<gg, 256, 0, stream>>>(ctx, Wo, bo, out);
}
